// WordDecoder_44092134261477
// MI455X (gfx1250) — compile-verified, matrix operand order checked by reading, not run on hardware
//
#include <hip/hip_runtime.h>
#include <hip/hip_bf16.h>
#include <math.h>

// Problem sizes (from reference): B=8, T=128, NS=12, S=50, PH=10, H=512, E=300, V=50000
#define BB 8
#define TT 128
#define NSNS 12
#define SS 50
#define PHPH 10
#define HH 512
#define EE 300
#define VV 50000

typedef __attribute__((ext_vector_type(16))) __bf16 bf16x16;
typedef __attribute__((ext_vector_type(8)))  float  f32x8;
typedef __attribute__((ext_vector_type(4)))  float  f32x4;

__device__ __forceinline__ float sigf(float x){ return 1.0f/(1.0f+__expf(-x)); }

__device__ __forceinline__ bf16x16 frag_zero(){
  bf16x16 r;
#pragma unroll
  for (int i=0;i<16;++i) r[i] = (__bf16)0.0f;
  return r;
}

// Unconditional vector load of one lane's 16 K-elements (K fully in-bounds).
// Layout (ISA 7.12.2, 16-bit A/B 16x32): lane carries K = kb..kb+7 and kb+16..kb+23.
__device__ __forceinline__ bf16x16 frag_vec(const float* __restrict__ p){
  f32x4 a = *(const f32x4*)(p);
  f32x4 b = *(const f32x4*)(p + 4);
  f32x4 c = *(const f32x4*)(p + 16);
  f32x4 d = *(const f32x4*)(p + 20);
  bf16x16 r;
  r[0]=(__bf16)a.x; r[1]=(__bf16)a.y; r[2]=(__bf16)a.z; r[3]=(__bf16)a.w;
  r[4]=(__bf16)b.x; r[5]=(__bf16)b.y; r[6]=(__bf16)b.z; r[7]=(__bf16)b.w;
  r[8]=(__bf16)c.x; r[9]=(__bf16)c.y; r[10]=(__bf16)c.z; r[11]=(__bf16)c.w;
  r[12]=(__bf16)d.x; r[13]=(__bf16)d.y; r[14]=(__bf16)d.z; r[15]=(__bf16)d.w;
  return r;
}

// Guarded scalar load for the single K-tail chunk (p = row base, stride 1 in K).
__device__ __forceinline__ bf16x16 frag_guard(const float* __restrict__ p, int kbase, int kreal){
  bf16x16 r;
#pragma unroll
  for (int e=0;e<8;++e){
    int k1 = kbase + e, k2 = kbase + 16 + e;
    r[e]   = (__bf16)((k1 < kreal) ? p[k1] : 0.0f);
    r[e+8] = (__bf16)((k2 < kreal) ? p[k2] : 0.0f);
  }
  return r;
}

// K-strided loads (B stored [K x N] row-major, walking one column).
__device__ __forceinline__ bf16x16 frag_strided(const float* __restrict__ p, long ld, int kbase){
  bf16x16 r;
#pragma unroll
  for (int e=0;e<8;++e){
    r[e]   = (__bf16)p[(long)(kbase + e) * ld];
    r[e+8] = (__bf16)p[(long)(kbase + 16 + e) * ld];
  }
  return r;
}
__device__ __forceinline__ bf16x16 frag_strided_guard(const float* __restrict__ p, long ld,
                                                      int kbase, int kreal){
  bf16x16 r;
#pragma unroll
  for (int e=0;e<8;++e){
    int k1 = kbase + e, k2 = kbase + 16 + e;
    r[e]   = (__bf16)((k1 < kreal) ? p[(long)k1*ld] : 0.0f);
    r[e+8] = (__bf16)((k2 < kreal) ? p[(long)k2*ld] : 0.0f);
  }
  return r;
}

// Generic C = act( A @ B(+T) + bias1 + bias2 + addsrc ), f32 in/out, bf16 WMMA core.
// bt=1: B stored [N x K] row-major (weight W, compute A@W^T). bt=0: B stored [K x N].
// Each wave owns one 16(M) x 64(N) strip: the A fragment is loaded once per K-chunk
// and reused by 4 WMMAs. Main K-loop is fully branch-free: out-of-range rows/cols are
// CLAMPED (their products land in C rows/cols that the epilogue never stores), so only
// the K tail needs guards. This keeps EXEC untouched in the loop (WMMA needs EXEC=all-1s).
__global__ void gemm_bf16_wmma(const float* __restrict__ A, int lda,
                               const float* __restrict__ Bm, int ldb, int bt,
                               float* __restrict__ C, int ldc,
                               const float* __restrict__ bias1,
                               const float* __restrict__ bias2,
                               const float* __restrict__ addsrc, long addStride, int addRows,
                               int M, int N, int Kreal, int act)
{
  const int lane = threadIdx.x & 31;
  const int wv   = threadIdx.x >> 5;
  const int ntg = (N + 63) >> 6;       // groups of 4 N-tiles
  const int mt  = (M + 15) >> 4;
  long g = (long)blockIdx.x * 8 + wv;
  if (g >= (long)ntg * mt) return;     // whole-wave exit
  const int m0 = (int)(g / ntg) * 16;
  const int n0 = (int)(g % ntg) * 64;

  const int arow = min(m0 + (lane & 15), M - 1);     // clamp: row>=M never stored
  const int klo  = (lane >> 4) << 3;   // lanes 16..31 carry the K+8 / K+24 halves
  const float* ap = A + (long)arow * lda;

  int col[4]; const float* bp[4];
#pragma unroll
  for (int j = 0; j < 4; ++j){
    col[j] = n0 + j * 16 + (lane & 15);
    int cc = min(col[j], N - 1);                     // clamp: col>=N never stored
    bp[j]  = bt ? (Bm + (long)cc * ldb) : (Bm + cc);
  }

  f32x8 acc[4];
#pragma unroll
  for (int j = 0; j < 4; ++j) acc[j] = (f32x8){0.f,0.f,0.f,0.f,0.f,0.f,0.f,0.f};

  const int kmain = Kreal & ~31;       // full 32-wide chunks: unconditional vector path
  if (bt){
    for (int k = 0; k < kmain; k += 32){
      bf16x16 af = frag_vec(ap + k + klo);
#pragma unroll
      for (int j = 0; j < 4; ++j)
        __builtin_prefetch(bp[j] + k + 32 + klo, 0, 1);   // speculative, safe past end
#pragma unroll
      for (int j = 0; j < 4; ++j){
        bf16x16 bf = frag_vec(bp[j] + k + klo);
        acc[j] = __builtin_amdgcn_wmma_f32_16x16x32_bf16(false, af, false, bf,
                                                         (short)0, acc[j], false, false);
      }
    }
  } else {
    for (int k = 0; k < kmain; k += 32){
      bf16x16 af = frag_vec(ap + k + klo);
#pragma unroll
      for (int j = 0; j < 4; ++j){
        bf16x16 bf = frag_strided(bp[j], ldb, k + klo);
        acc[j] = __builtin_amdgcn_wmma_f32_16x16x32_bf16(false, af, false, bf,
                                                         (short)0, acc[j], false, false);
      }
    }
  }
  if (kmain < Kreal){                  // single guarded tail chunk
    bf16x16 af = frag_guard(ap, kmain + klo, Kreal);
#pragma unroll
    for (int j = 0; j < 4; ++j){
      bf16x16 bf = bt ? frag_guard(bp[j], kmain + klo, Kreal)
                      : frag_strided_guard(bp[j], ldb, kmain + klo, Kreal);
      acc[j] = __builtin_amdgcn_wmma_f32_16x16x32_bf16(false, af, false, bf,
                                                       (short)0, acc[j], false, false);
    }
  }

  // C layout (ISA 7.12.2): VGPR r -> M = r (lanes 0..15) / 8+r (lanes 16..31); N = lane&15
  const int rbase = (lane >> 4) * 8;
#pragma unroll
  for (int j = 0; j < 4; ++j){
    if (col[j] >= N) continue;
    const int cc = col[j];
    float badd = 0.0f;
    if (bias1) badd += bias1[cc];
    if (bias2) badd += bias2[cc];
#pragma unroll
    for (int r = 0; r < 8; ++r){
      int row = m0 + r + rbase;
      if (row < M){
        float v = acc[j][r] + badd;
        if (addsrc && row < addRows) v += addsrc[(long)row * addStride + cc];
        if (act == 1) v = tanhf(v);
        C[(long)row * ldc + cc] = v;
      }
    }
  }
}

// ---------- supporting elementwise / fused kernels ----------

// Build concat input [1024 x 832]: [emb(300) | mask*spo_gathered(512) | 0...]
__global__ void gather_concat(const float* __restrict__ emb, const float* __restrict__ spo,
                              const int* __restrict__ sid, const float* __restrict__ smask,
                              float* __restrict__ out){
  long i = (long)blockIdx.x * blockDim.x + threadIdx.x;
  if (i >= 1024L * 832) return;
  int row = (int)(i / 832), col = (int)(i % 832);
  float v;
  if (col < EE) v = emb[(long)row * EE + col];
  else if (col < EE + HH){
    int s = sid[row];
    v = smask[row] * spo[((long)((row >> 7) * NSNS + s)) * HH + (col - EE)];
  } else v = 0.0f;
  out[i] = v;
}

// Pack [W_word | W_plan] into one [300 x 832] weight so premix is a single GEMM.
__global__ void pack_premix_w(const float* __restrict__ Ww, const float* __restrict__ Wp,
                              float* __restrict__ Wc){
  long i = (long)blockIdx.x * blockDim.x + threadIdx.x;
  if (i >= 300L * 832) return;
  int e = (int)(i / 832), c = (int)(i % 832);
  Wc[i] = (c < EE) ? Ww[(long)e * EE + c]
        : ((c < EE + HH) ? Wp[(long)e * HH + (c - EE)] : 0.0f);
}

__global__ void init_state(const float* __restrict__ h0, const float* __restrict__ c0,
                           float* __restrict__ h_0, float* __restrict__ c_0,
                           float* __restrict__ h_1, float* __restrict__ c_1){
  int i = blockIdx.x * blockDim.x + threadIdx.x;
  if (i >= 16 * HH) return;
  int b = i >> 9;
  bool ok = (b < BB);                 // rows 8..15 are WMMA M-padding, keep zero
  h_0[i] = ok ? h0[i] : 0.0f;
  c_0[i] = ok ? c0[i] : 0.0f;
  h_1[i] = ok ? h0[BB * HH + i] : 0.0f;
  c_1[i] = ok ? c0[BB * HH + i] : 0.0f;
}

// gates g (16x2048, rows 0..7 valid) -> c,h update, write y[(b*T+t)]
__global__ void lstm_update(const float* __restrict__ g, float* __restrict__ h,
                            float* __restrict__ c, float* __restrict__ y, int t){
  int i = blockIdx.x * blockDim.x + threadIdx.x;
  if (i >= BB * HH) return;
  int b = i >> 9, j = i & (HH - 1);
  const float* gr = g + (long)b * 4 * HH;
  float gi = gr[j], gf = gr[HH + j], gg = gr[2 * HH + j], go = gr[3 * HH + j];
  float c2 = sigf(gf) * c[i] + sigf(gi) * tanhf(gg);
  float h2 = sigf(go) * tanhf(c2);
  c[i] = c2; h[i] = h2;
  y[((long)(b * TT + t)) * HH + j] = h2;
}

__global__ void copy_state_out(const float* __restrict__ h_0, const float* __restrict__ h_1,
                               const float* __restrict__ c_0, const float* __restrict__ c_1,
                               float* __restrict__ out){
  int i = blockIdx.x * blockDim.x + threadIdx.x;
  if (i >= 2 * BB * HH) return;
  int l = i >> 12, r = i & 4095;
  out[i]              = (l ? h_1 : h_0)[r];   // hT
  out[2*BB*HH + i]    = (l ? c_1 : c_0)[r];   // cT
}

// Per token: masked softmax attention over S=50, context c, concat [c|h] buffer, p_gen.
__global__ void attn_ctx_pgen(const float* __restrict__ q, const float* __restrict__ mem,
                              const int* __restrict__ mlen, const float* __restrict__ y1,
                              const float* __restrict__ emb,
                              const float* __restrict__ Wpgc, const float* __restrict__ bpgc,
                              const float* __restrict__ Wpgz, const float* __restrict__ Wpgy,
                              float* __restrict__ enc_out, float* __restrict__ concat2,
                              float* __restrict__ pgen)
{
  const int row = blockIdx.x;
  const int b = row >> 7;
  const int tid = threadIdx.x, lane = tid & 31, wv = tid >> 5;
  __shared__ float sq[EE];
  __shared__ float sa[SS];
  __shared__ float sred[8];
  __shared__ float sscal;
  for (int e = tid; e < EE; e += 256) sq[e] = q[(long)row * EE + e];
  __syncthreads();
  for (int s = wv; s < SS; s += 8){
    const float* mr = mem + (long)(b * SS + s) * EE;
    float p = 0.0f;
    for (int e = lane; e < EE; e += 32) p += sq[e] * mr[e];
    for (int off = 16; off; off >>= 1) p += __shfl_down(p, off, 32);
    if (lane == 0) sa[s] = (s < mlen[b]) ? p : -1e9f;
  }
  __syncthreads();
  if (tid == 0){ float m = -3e38f; for (int s = 0; s < SS; ++s) m = fmaxf(m, sa[s]); sscal = m; }
  __syncthreads();
  float mx = sscal;
  __syncthreads();
  if (tid < SS) sa[tid] = __expf(sa[tid] - mx);
  __syncthreads();
  if (tid == 0){ float t = 0.0f; for (int s = 0; s < SS; ++s) t += sa[s]; sscal = t; }
  __syncthreads();
  if (tid < SS){ sa[tid] /= sscal; enc_out[(long)row * SS + tid] = sa[tid]; }
  __syncthreads();
  float part = 0.0f;
  for (int e = tid; e < 832; e += 256){
    float v;
    if (e < EE){
      const float* mb = mem + (long)b * SS * EE + e;
      float acc = 0.0f;
      for (int s = 0; s < SS; ++s) acc += sa[s] * mb[(long)s * EE];
      v = acc; part += acc * Wpgc[e];
    } else if (e < EE + HH){
      float hv = y1[(long)row * HH + (e - EE)];
      v = hv; part += hv * Wpgz[e - EE];
    } else v = 0.0f;
    concat2[(long)row * 832 + e] = v;
  }
  for (int e = tid; e < EE; e += 256) part += emb[(long)row * EE + e] * Wpgy[e];
  for (int off = 16; off; off >>= 1) part += __shfl_down(part, off, 32);
  if (lane == 0) sred[wv] = part;
  __syncthreads();
  if (tid == 0){
    float t = bpgc[0];
    for (int w = 0; w < 8; ++w) t += sred[w];
    pgen[row] = 1.0f / (1.0f + __expf(-t));
  }
}

// Row softmax over V=50000, scaled by p_gen, written into vocab_pred_dist region.
__global__ void softmax_pgen_scale(const float* __restrict__ ro, const float* __restrict__ pgen,
                                   float* __restrict__ vp){
  const int row = blockIdx.x;
  const float* x = ro + (long)row * VV;
  float* o = vp + (long)row * VV;
  const int tid = threadIdx.x, lane = tid & 31, wv = tid >> 5;
  __shared__ float sh[8];
  float m = -3e38f;
  for (int v = tid; v < VV; v += 256) m = fmaxf(m, x[v]);
  for (int off = 16; off; off >>= 1) m = fmaxf(m, __shfl_down(m, off, 32));
  if (lane == 0) sh[wv] = m;
  __syncthreads();
  if (tid == 0){ float t = sh[0]; for (int w = 1; w < 8; ++w) t = fmaxf(t, sh[w]); sh[0] = t; }
  __syncthreads();
  const float mx = sh[0];
  __syncthreads();
  float s = 0.0f;
  for (int v = tid; v < VV; v += 256) s += __expf(x[v] - mx);
  for (int off = 16; off; off >>= 1) s += __shfl_down(s, off, 32);
  if (lane == 0) sh[wv] = s;
  __syncthreads();
  if (tid == 0){ float t = 0.0f; for (int w = 0; w < 8; ++w) t += sh[w]; sh[0] = t; }
  __syncthreads();
  const float k = pgen[row] / sh[0];
  for (int v = tid; v < VV; v += 256) o[v] = __expf(x[v] - mx) * k;
}

// Pointer-generator scatter: vp[row, ids[b,s,ph]] += (1-pg)*attn[row,s]/len(b,s)*mask
__global__ void scatter_copy(const float* __restrict__ enc, const float* __restrict__ pgen,
                             const int* __restrict__ ids, const float* __restrict__ pmask,
                             float* __restrict__ vp){
  int row = blockIdx.x; int b = row >> 7;
  int i = threadIdx.x; if (i >= SS * PHPH) return;
  int s = i / PHPH, ph = i % PHPH;
  const float* mk = pmask + (long)(b * SS + s) * PHPH;
  float len = 1e-8f;
#pragma unroll
  for (int p = 0; p < PHPH; ++p) len += mk[p];
  float a = enc[(long)row * SS + s] * (1.0f - pgen[row]) / len;
  float val = a * mk[ph];
  int id = ids[(long)(b * SS + s) * PHPH + ph];
  atomicAdd(vp + (long)row * VV + id, val);
}

// ---------- host side ----------

static inline void launch_gemm(hipStream_t st, const float* A, int lda,
                               const float* Bm, int ldb, int bt,
                               float* C, int ldc, const float* b1, const float* b2,
                               const float* add, long addStride, int addRows,
                               int M, int N, int Kr, int act){
  long ntg = (N + 63) / 64, mt = (M + 15) / 16;
  long tiles = ntg * mt;
  int blocks = (int)((tiles + 7) / 8);
  gemm_bf16_wmma<<<blocks, 256, 0, st>>>(A, lda, Bm, ldb, bt, C, ldc, b1, b2,
                                         add, addStride, addRows, M, N, Kr, act);
}

extern "C" void kernel_launch(void* const* d_in, const int* in_sizes, int n_in,
                              void* d_out_, int out_size, void* d_ws, size_t ws_size,
                              hipStream_t stream)
{
  const float* emb   = (const float*)d_in[0];
  const float* spo   = (const float*)d_in[1];
  const int*   sid   = (const int*)  d_in[2];
  const float* smask = (const float*)d_in[3];
  const float* mem   = (const float*)d_in[4];
  const int*   mlen  = (const int*)  d_in[5];
  const int*   phids = (const int*)  d_in[6];
  const float* phmsk = (const float*)d_in[7];
  const float* Wword = (const float*)d_in[8];
  const float* bword = (const float*)d_in[9];
  const float* Wplan = (const float*)d_in[10];
  const float* bplan = (const float*)d_in[11];
  const float* Wih0  = (const float*)d_in[12];
  const float* Whh0  = (const float*)d_in[13];
  const float* bih0  = (const float*)d_in[14];
  const float* bhh0  = (const float*)d_in[15];
  const float* Wih1  = (const float*)d_in[16];
  const float* Whh1  = (const float*)d_in[17];
  const float* bih1  = (const float*)d_in[18];
  const float* bhh1  = (const float*)d_in[19];
  const float* Wain  = (const float*)d_in[20];
  const float* Waout = (const float*)d_in[21];
  const float* Wread = (const float*)d_in[22];
  const float* bread = (const float*)d_in[23];
  const float* Wpgc  = (const float*)d_in[24];
  const float* bpgc  = (const float*)d_in[25];
  const float* Wpgz  = (const float*)d_in[26];
  const float* Wpgy  = (const float*)d_in[27];
  const float* h0    = (const float*)d_in[28];
  const float* c0    = (const float*)d_in[29];

  float* out = (float*)d_out_;
  float* ws  = (float*)d_ws;

  // output layout (return order, flat): hT | cT | enc_attn | vocab_pred_dist | readouts
  float* outHT  = out;                       // 2*8*512   = 8192
  float* outENC = out + 16384;               // 8*128*50  = 51200
  float* outVP  = out + 67584;               // 8*128*50000
  float* outRO  = out + 51267584L;           // 8*128*50000

  long o = 0;
  float* concat_in = ws + o; o += 1024L * 832;   // [emb|spo] per token
  float* Wc        = ws + o; o += 300L  * 832;   // packed premix weight
  float* rnn_in    = ws + o; o += 1024L * 300;
  float* gih       = ws + o; o += 1024L * 2048;  // precomputed input gates (per layer, reused)
  float* gbuf      = ws + o; o += 16L   * 2048;  // per-step gates (M padded to 16)
  float* hcur0     = ws + o; o += 16L * 512;
  float* ccur0     = ws + o; o += 16L * 512;
  float* hcur1     = ws + o; o += 16L * 512;
  float* ccur1     = ws + o; o += 16L * 512;
  float* y0        = ws + o; o += 1024L * 512;
  float* y1        = ws + o; o += 1024L * 512;
  float* qbuf      = ws + o; o += 1024L * 300;
  float* concat2   = ws + o; o += 1024L * 832;   // [ctx|h] per token
  float* decbuf    = ws + o; o += 1024L * 512;
  float* pgen      = ws + o; o += 1024;

  // 1) gather + premix (tanh(emb@Ww^T + spo@Wp^T + b)) as one WMMA GEMM
  gather_concat<<<(1024 * 832 + 255) / 256, 256, 0, stream>>>(emb, spo, sid, smask, concat_in);
  pack_premix_w<<<(300 * 832 + 255) / 256, 256, 0, stream>>>(Wword, Wplan, Wc);
  launch_gemm(stream, concat_in, 832, Wc, 832, 1, rnn_in, 300,
              bword, bplan, nullptr, 0, 0, 1024, 300, 812, 1);

  // 2) LSTM layer 0: precompute x@Wih^T + bih + bhh for all t, then sequential recurrence
  launch_gemm(stream, rnn_in, 300, Wih0, 300, 1, gih, 2048,
              bih0, bhh0, nullptr, 0, 0, 1024, 2048, 300, 0);
  init_state<<<32, 256, 0, stream>>>(h0, c0, hcur0, ccur0, hcur1, ccur1);
  for (int t = 0; t < TT; ++t){
    launch_gemm(stream, hcur0, 512, Whh0, 512, 1, gbuf, 2048,
                nullptr, nullptr, gih + (long)t * 2048, (long)TT * 2048, BB,
                16, 2048, 512, 0);
    lstm_update<<<16, 256, 0, stream>>>(gbuf, hcur0, ccur0, y0, t);
  }

  // 3) LSTM layer 1
  launch_gemm(stream, y0, 512, Wih1, 512, 1, gih, 2048,
              bih1, bhh1, nullptr, 0, 0, 1024, 2048, 512, 0);
  for (int t = 0; t < TT; ++t){
    launch_gemm(stream, hcur1, 512, Whh1, 512, 1, gbuf, 2048,
                nullptr, nullptr, gih + (long)t * 2048, (long)TT * 2048, BB,
                16, 2048, 512, 0);
    lstm_update<<<16, 256, 0, stream>>>(gbuf, hcur1, ccur1, y1, t);
  }
  copy_state_out<<<32, 256, 0, stream>>>(hcur0, hcur1, ccur0, ccur1, outHT);

  // 4) attention: q = y1 @ W_attn_in  (B stored [K x N], bt=0)
  launch_gemm(stream, y1, 512, Wain, 300, 0, qbuf, 300,
              nullptr, nullptr, nullptr, 0, 0, 1024, 300, 512, 0);
  attn_ctx_pgen<<<1024, 256, 0, stream>>>(qbuf, mem, mlen, y1, emb,
                                          Wpgc, bpgc, Wpgz, Wpgy,
                                          outENC, concat2, pgen);

  // 5) dec_outs = tanh([c|h] @ W_attn_out^T)
  launch_gemm(stream, concat2, 832, Waout, 812, 1, decbuf, 512,
              nullptr, nullptr, nullptr, 0, 0, 1024, 512, 812, 1);

  // 6) readouts = dec_outs @ W_read^T + b_read  (the 52 GFLOP GEMM, N=50000)
  launch_gemm(stream, decbuf, 512, Wread, 512, 1, outRO, VV,
              bread, nullptr, nullptr, 0, 0, 1024, VV, 512, 0);

  // 7) vocab_pred_dist = p_gen * softmax(readouts), then pointer-gen scatter-add
  softmax_pgen_scale<<<1024, 256, 0, stream>>>(outRO, pgen, outVP);
  scatter_copy<<<1024, 512, 0, stream>>>(outENC, pgen, phids, phmsk, outVP);
}
